// RAE_60112362275063
// MI455X (gfx1250) — matrix operand order repaired and checked by reading, not compiled, not run on hardware
//
#include <hip/hip_runtime.h>

#define XD 38
#define XP 64          // x K padded to 2 wmma k-slabs
#define HD 128
#define GD 384         // 3*H
#define TSEQ 128
#define NBATCH 2048
#define BT 16          // batch rows per block (one WMMA M tile)
#define NW 8           // waves per block: wave w owns columns [16w,16w+16) of r/z/n/h
#define NTHREADS (NW * 32)

// padded LDS strides (all odd multiples of 16B -> conflict-free b128 frag loads)
#define WS 392         // Whh^T / Wih^T row stride in halves (GD + 8)
#define LS 56          // lin^T row stride in halves (48 + 8)
#define HS 136         // h tile stride in halves (HD + 8)
#define XS 72          // x tile stride in halves (XP + 8)

typedef __attribute__((ext_vector_type(16))) _Float16 v16h;
typedef __attribute__((ext_vector_type(8)))  _Float16 v8h;
typedef __attribute__((ext_vector_type(8)))  float    v8f;

__device__ __forceinline__ v8f wmma_f16(v16h a, v16h b, v8f c) {
  // D = A(16x32 f16) * B(32x16 f16) + C(16x16 f32)
  return __builtin_amdgcn_wmma_f32_16x16x32_f16(false, a, false, b, (short)0, c,
                                                false, false);
}

// A fragment: lane holds row m = lane&15; K chunks {kh..kh+7} U {kh+16..kh+23},
// kh = (lane>>4)*8, within the 32-wide k-slab starting at k0.
__device__ __forceinline__ v16h afrag(const _Float16* base, int stride, int m,
                                      int kh, int k0) {
  const _Float16* p = base + m * stride + k0 + kh;
  v8h lo = *(const v8h*)p;
  v8h hi = *(const v8h*)(p + 16);
  return __builtin_shufflevector(lo, hi, 0, 1, 2, 3, 4, 5, 6, 7, 8, 9, 10, 11,
                                 12, 13, 14, 15);
}

// B fragment: lane holds column N = n0+(lane&15); K chunks {kh..kh+7} U {kh+16..kh+23}, kh = (lane>>4)*8.
// Weights stored K-major in LDS: W[k][n].
__device__ __forceinline__ v16h bfrag(const _Float16* base, int stride, int lane,
                                      int k0, int n0) {
  const _Float16* p = base + (k0 + ((lane >> 4) & 1) * 8) * stride + n0 + (lane & 15);
  v16h v;
#pragma unroll
  for (int i = 0; i < 8; ++i) {
    v[i] = p[i * stride];
    v[8 + i] = p[(16 + i) * stride];
  }
  return v;
}

// Single v_rcp_f32 instead of IEEE divide chains.
__device__ __forceinline__ float fast_rcp(float x) {
  return __builtin_amdgcn_rcpf(x);
}
__device__ __forceinline__ float sigmoidf_fast(float x) {
  return fast_rcp(1.0f + __expf(-x));   // v_exp + v_rcp
}
__device__ __forceinline__ float tanhf_fast(float x) {
  return 1.0f - 2.0f * fast_rcp(1.0f + __expf(2.0f * x));  // v_exp + v_rcp + fma
}

__global__ __launch_bounds__(NTHREADS) void rae_gru_kernel(
    const float* __restrict__ x,
    const float* __restrict__ eWih, const float* __restrict__ eWhh,
    const float* __restrict__ eBih, const float* __restrict__ eBhh,
    const float* __restrict__ dWih, const float* __restrict__ dWhh,
    const float* __restrict__ dBih, const float* __restrict__ dBhh,
    const float* __restrict__ linW, const float* __restrict__ linB,
    float* __restrict__ out) {
  __shared__ __align__(16) _Float16 sWhh[HD][WS];   // W^T: [k][n], n in [0,384)
  __shared__ __align__(16) _Float16 sWih[XP][WS];   // W^T: [k][n], k padded 38->64
  __shared__ __align__(16) _Float16 sLin[HD][LS];   // lin^T: [k][n], n padded 38->48
  __shared__ __align__(16) _Float16 sHbuf[2][BT][HS]; // hidden state ping-pong
  __shared__ __align__(16) _Float16 sXbuf[2][BT][XS]; // x_t / o_{t-1} ping-pong
  __shared__ float sBih[GD];
  __shared__ float sBhh[GD];
  __shared__ float sLb[48];

  const int tid  = threadIdx.x;
  const int lane = tid & 31;
  const int w    = tid >> 5;
  const int nc   = lane & 15;                 // accumulator column within tile
  const int m16  = lane & 15;                 // A-frag row
  const int kh   = ((lane >> 4) & 1) * 8;     // A-frag k chunk base
  const int mr0  = ((lane >> 4) & 1) * 8;     // accumulator M base
  const int bbase = blockIdx.x * BT;

  // Column slice owned by this wave (same slice for r, z, n, h).
  const int NC0 = w * 16;                     // hidden column base
  const int col = NC0 + nc;                   // this lane's hidden column

  // Ping-pong pointers: gru_step reads *r, writes sHw.
  _Float16 (*sHr)[HS] = sHbuf[0];
  _Float16 (*sHw)[HS] = sHbuf[1];
  _Float16 (*sXr)[XS] = sXbuf[0];
  _Float16 (*sXw)[XS] = sXbuf[1];

  // Persistent per-lane state in accumulator layout: element (mr0+r, col).
  float hreg[8];
#pragma unroll
  for (int r = 0; r < 8; ++r) hreg[r] = 0.0f;

  // Loop-invariant weight B-fragments, hoisted to registers per phase.
  v16h bR[6], bZ[6], bN[6];   // [0..1]=Wih slabs, [2..5]=Whh slabs (bN: 0..3 Whh, 4..5 Wih)
  float biasR, biasZ, bgi, bgh;

  auto hoist_weights = [&]() {
#pragma unroll
    for (int s = 0; s < 2; ++s) {
      bR[s] = bfrag(&sWih[0][0], WS, lane, s * 32, NC0);
      bZ[s] = bfrag(&sWih[0][0], WS, lane, s * 32, 128 + NC0);
      bN[4 + s] = bfrag(&sWih[0][0], WS, lane, s * 32, 256 + NC0);
    }
#pragma unroll
    for (int s = 0; s < 4; ++s) {
      bR[2 + s] = bfrag(&sWhh[0][0], WS, lane, s * 32, NC0);
      bZ[2 + s] = bfrag(&sWhh[0][0], WS, lane, s * 32, 128 + NC0);
      bN[s] = bfrag(&sWhh[0][0], WS, lane, s * 32, 256 + NC0);
    }
    biasR = sBih[col] + sBhh[col];
    biasZ = sBih[128 + col] + sBhh[128 + col];
    bgi   = sBih[256 + col];
    bgh   = sBhh[256 + col];
  };

  // Encoder x staging: fixed per-thread element slots (BT*XD = 608 <= 3*256).
  const int i0 = tid, i1 = tid + NTHREADS, i2 = tid + 2 * NTHREADS;
  const int xm0 = i0 / XD, xc0 = i0 % XD;
  const int xm1 = i1 / XD, xc1 = i1 % XD;
  const bool has2 = (i2 < BT * XD);
  const int xm2 = has2 ? i2 / XD : 0, xc2 = has2 ? i2 % XD : 0;
  float xv0 = 0.f, xv1 = 0.f, xv2 = 0.f;
  auto ldx = [&](int t) {  // global -> registers (issued early, latency hidden)
    xv0 = x[((size_t)(bbase + xm0) * TSEQ + t) * XD + xc0];
    xv1 = x[((size_t)(bbase + xm1) * TSEQ + t) * XD + xc1];
    if (has2) xv2 = x[((size_t)(bbase + xm2) * TSEQ + t) * XD + xc2];
  };
  auto stx = [&](_Float16 (*dst)[XS]) {  // registers -> LDS (f32->f16)
    dst[xm0][xc0] = (_Float16)xv0;
    dst[xm1][xc1] = (_Float16)xv1;
    if (has2) dst[xm2][xc2] = (_Float16)xv2;
  };

  // ---- init: zero both h and x buffers (padding columns stay zero) ----
  for (int i = tid; i < 2 * BT * HS; i += NTHREADS)
    (&sHbuf[0][0][0])[i] = (_Float16)0.f;
  for (int i = tid; i < 2 * BT * XS; i += NTHREADS)
    (&sXbuf[0][0][0])[i] = (_Float16)0.f;

  // ---- load encoder weights (transposed to K-major, f32->f16) ----
  for (int i = tid; i < GD * HD; i += NTHREADS) {
    int n = i / HD, k = i % HD;
    sWhh[k][n] = (_Float16)eWhh[i];
  }
  for (int i = tid; i < GD * XP; i += NTHREADS) {
    int n = i / XP, k = i % XP;
    sWih[k][n] = (k < XD) ? (_Float16)eWih[n * XD + k] : (_Float16)0.f;
  }
  for (int i = tid; i < GD; i += NTHREADS) {
    sBih[i] = eBih[i];
    sBhh[i] = eBhh[i];
  }
  ldx(0);
  stx(sXr);
  __syncthreads();
  hoist_weights();

  // One fused GRU step: reads sXr/sHr, writes sHw (no read/write overlap).
  // Per-step LDS traffic: 12x ds_load_b128 (A frags) + 8x ds_store_b16 (h).
  auto gru_step = [&]() {
    v16h ax[2], ah[4];
#pragma unroll
    for (int s = 0; s < 2; ++s) ax[s] = afrag(&sXr[0][0], XS, m16, kh, s * 32);
#pragma unroll
    for (int s = 0; s < 4; ++s) ah[s] = afrag(&sHr[0][0], HS, m16, kh, s * 32);

    v8f accR = {};
#pragma unroll
    for (int s = 0; s < 2; ++s) accR = wmma_f16(ax[s], bR[s], accR);
#pragma unroll
    for (int s = 0; s < 4; ++s) accR = wmma_f16(ah[s], bR[2 + s], accR);

    v8f accZ = {};
#pragma unroll
    for (int s = 0; s < 2; ++s) accZ = wmma_f16(ax[s], bZ[s], accZ);
#pragma unroll
    for (int s = 0; s < 4; ++s) accZ = wmma_f16(ah[s], bZ[2 + s], accZ);

    v8f agh = {};
#pragma unroll
    for (int s = 0; s < 4; ++s) agh = wmma_f16(ah[s], bN[s], agh);
    v8f agi = {};
#pragma unroll
    for (int s = 0; s < 2; ++s) agi = wmma_f16(ax[s], bN[4 + s], agi);

#pragma unroll
    for (int r = 0; r < 8; ++r) {
      const int M = mr0 + r;
      const float rv = sigmoidf_fast(accR[r] + biasR);
      const float zv = sigmoidf_fast(accZ[r] + biasZ);
      const float nv = tanhf_fast((agi[r] + bgi) + rv * (agh[r] + bgh));
      const float hn = (1.0f - zv) * nv + zv * hreg[r];
      hreg[r] = hn;
      sHw[M][col] = (_Float16)hn;
    }
  };

  // ===================== encoder: 1 barrier per step =====================
  for (int t = 0; t < TSEQ; ++t) {
    if (t + 1 < TSEQ) ldx(t + 1);   // issue global loads early
    gru_step();                      // reads sXr/sHr, writes sHw
    if (t + 1 < TSEQ) stx(sXw);      // stage next x into write buffer
    __syncthreads();                 // publish sHw and sXw
    { _Float16 (*tmp)[HS] = sHr; sHr = sHw; sHw = tmp; }
    { _Float16 (*tmp)[XS] = sXr; sXr = sXw; sXw = tmp; }
  }

  // ---- phase switch: load decoder weights, lin, zero o_prev ----
  for (int i = tid; i < GD * HD; i += NTHREADS) {
    int n = i / HD, k = i % HD;
    sWhh[k][n] = (_Float16)dWhh[i];
  }
  for (int i = tid; i < GD * XP; i += NTHREADS) {
    int n = i / XP, k = i % XP;
    sWih[k][n] = (k < XD) ? (_Float16)dWih[n * XD + k] : (_Float16)0.f;
  }
  for (int i = tid; i < GD; i += NTHREADS) {
    sBih[i] = dBih[i];
    sBhh[i] = dBhh[i];
  }
  for (int i = tid; i < HD * 48; i += NTHREADS) {
    int k = i / 48, n = i % 48;
    sLin[k][n] = (n < XD) ? (_Float16)linW[n * HD + k] : (_Float16)0.f;
  }
  for (int i = tid; i < 48; i += NTHREADS) sLb[i] = (i < XD) ? linB[i] : 0.f;
  for (int i = tid; i < BT * XS; i += NTHREADS)
    (&sXr[0][0])[i] = (_Float16)0.f;   // o_prev = GO = zeros
  __syncthreads();
  hoist_weights();       // re-hoist decoder weight fragments

  // ===================== decoder (autoregressive) =====================
  for (int t = 0; t < TSEQ; ++t) {
    gru_step();          // reads sXr (= o_{t-1}) and sHr, writes sHw
    __syncthreads();     // all h columns of sHw visible

    // projection: o = h_new @ linW^T + b ; 3 N-tiles on waves 0..2
    if (w < 3) {
      const int N0 = w * 16;
      v8f acc = {};
#pragma unroll
      for (int s = 0; s < 4; ++s) {
        v16h a = afrag(&sHw[0][0], HS, m16, kh, s * 32);
        acc = wmma_f16(a, bfrag(&sLin[0][0], LS, lane, s * 32, N0), acc);
      }
      const int pc = N0 + nc;
      const float b = sLb[pc];
#pragma unroll
      for (int r = 0; r < 8; ++r) {
        const int M = mr0 + r;
        const float val = acc[r] + b;
        sXr[M][pc] = (_Float16)val;  // padded cols (>=38) write exact 0
        if (pc < XD)
          out[((size_t)(bbase + M) * TSEQ + t) * XD + pc] = val;
      }
    }
    __syncthreads();     // o_t visible in sXr for next step
    { _Float16 (*tmp)[HS] = sHr; sHr = sHw; sHw = tmp; }
  }
}

extern "C" void kernel_launch(void* const* d_in, const int* in_sizes, int n_in,
                              void* d_out, int out_size, void* d_ws,
                              size_t ws_size, hipStream_t stream) {
  (void)in_sizes; (void)n_in; (void)out_size; (void)d_ws; (void)ws_size;
  const float* x    = (const float*)d_in[0];
  const float* eWih = (const float*)d_in[1];
  const float* eWhh = (const float*)d_in[2];
  const float* eBih = (const float*)d_in[3];
  const float* eBhh = (const float*)d_in[4];
  const float* dWih = (const float*)d_in[5];
  const float* dWhh = (const float*)d_in[6];
  const float* dBih = (const float*)d_in[7];
  const float* dBhh = (const float*)d_in[8];
  const float* linW = (const float*)d_in[9];
  const float* linB = (const float*)d_in[10];

  dim3 grid(NBATCH / BT);
  dim3 block(NTHREADS);
  rae_gru_kernel<<<grid, block, 0, stream>>>(x, eWih, eWhh, eBih, eBhh, dWih,
                                             dWhh, dBih, dBhh, linW, linB,
                                             (float*)d_out);
}
